// FourierEmbedding_83021717832062
// MI455X (gfx1250) — matrix operand order repaired and checked by reading, not compiled, not run on hardware
//
#include <hip/hip_runtime.h>
#include <hip/hip_bf16.h>
#include <math.h>

typedef __attribute__((ext_vector_type(16))) _Float16 v16h;
typedef __attribute__((ext_vector_type(8)))  float    v8f;

#define FE_D      4
#define FE_F      64
#define FE_H      128
#define FE_FIN    129
#define FE_TWO_PI 6.28318530717958647692f
#define FE_EPS    1e-5f

// Swizzled weight tiles: each tile is a 32x16 f16 B-fragment, 512 f16 = 1KB.
// Lane l of a wave holds 16 contiguous f16 (32B): elements [2v+j] = B[k][n]
// with n = l&15, k = (l>>4)*8 + 2v + j + (v>=4 ? 8 : 0)  (matches CDNA5 16-bit B layout).
#define T1_COUNT (FE_D * 5 * 8)   // w1 padded K=160: 5 K-chunks x 8 N-tiles x D
#define T2_COUNT (FE_D * 4 * 8)   // w2 K=128
#define T3_COUNT (4 * 8)          // w3 K=128
#define T2_BASE  T1_COUNT
#define T3_BASE  (T1_COUNT + T2_COUNT)
#define TILE_ELTS 512

// ---------------- prep: f32 weights -> f16 swizzled B-fragments ----------------
__global__ void fe_prep(const float* __restrict__ w1, const float* __restrict__ w2,
                        const float* __restrict__ w3, _Float16* __restrict__ ws) {
    const int t    = blockIdx.x;          // tile id, one wave per tile
    const int lane = threadIdx.x;         // 32 threads
    const int col  = lane & 15;
    const int g    = lane >> 4;

    int region, d = 0, kc, nt;
    if (t < T2_BASE)      { region = 0; d = t / 40;          int r = t % 40; kc = r / 8; nt = r % 8; }
    else if (t < T3_BASE) { region = 1; int u = t - T2_BASE; d = u / 32; kc = (u % 32) / 8; nt = u % 8; }
    else                  { region = 2; int u = t - T3_BASE; kc = u / 8; nt = u % 8; }

    const int h = nt * 16 + col;
    #pragma unroll
    for (int v = 0; v < 8; ++v) {
        #pragma unroll
        for (int j = 0; j < 2; ++j) {
            const int k = kc * 32 + g * 8 + 2 * v + j + ((v >= 4) ? 8 : 0);   // source K row
            float val = 0.0f;
            if (region == 0)      { if (k < FE_FIN) val = w1[(d * FE_FIN + k) * FE_H + h]; }
            else if (region == 1) { val = w2[(d * FE_H + k) * FE_H + h]; }
            else                  { val = w3[k * FE_H + h]; }
            ws[(size_t)t * TILE_ELTS + lane * 16 + 2 * v + j] = (_Float16)val;
        }
    }
}

// Fourier feature element for (padded) K index kk of dim d
__device__ __forceinline__ float fe_feat(int kk, float x, const float* fr) {
    if (kk < FE_F)      return __cosf(x * fr[kk] * FE_TWO_PI);
    if (kk < 2 * FE_F)  return __sinf(x * fr[kk - FE_F] * FE_TWO_PI);
    if (kk == 2 * FE_F) return x;
    return 0.0f;
}

// ---------------- main kernel: 8 waves, one 16x128 row-tile per wave ----------------
#define HB_STRIDE 136   // 128 + 8 f16 pad -> bank-conflict-free column reads

__global__ __launch_bounds__(256) void fe_main(
        const float* __restrict__ cin,  const float* __restrict__ freqs,
        const float* __restrict__ b1,   const float* __restrict__ g1, const float* __restrict__ be1,
        const float* __restrict__ b2,   const float* __restrict__ og, const float* __restrict__ ob,
        const float* __restrict__ b3,   const _Float16* __restrict__ ws,
        float* __restrict__ out) {
    __shared__ float    sfreq[FE_D * FE_F];
    __shared__ _Float16 hbuf[8][16][HB_STRIDE];

    const int tid  = threadIdx.x;
    const int lane = tid & 31;
    const int wave = tid >> 5;
    const int col  = lane & 15;
    const int g    = lane >> 4;

    if (tid < FE_D * FE_F) sfreq[tid] = freqs[tid];
    __syncthreads();

    const int row0 = (blockIdx.x * 8 + wave) * 16;

    float xin[FE_D];
    #pragma unroll
    for (int d = 0; d < FE_D; ++d) xin[d] = cin[(row0 + col) * FE_D + d];

    // emb accumulators (C layout: lane col, VGPR v -> row v+8g), 16x128 tile
    v8f acc2[8];
    #pragma unroll
    for (int nt = 0; nt < 8; ++nt)
        #pragma unroll
        for (int i = 0; i < 8; ++i) acc2[nt][i] = 0.0f;

    #pragma unroll 1
    for (int d = 0; d < FE_D; ++d) {
        const float* fr = sfreq + d * FE_F;
        const float  x  = xin[d];

        // --- GEMM1: feat[16x160] @ w1p[160x128], K-chunk outer (1 A frag live) ---
        v8f acc[8];
        #pragma unroll
        for (int nt = 0; nt < 8; ++nt)
            #pragma unroll
            for (int i = 0; i < 8; ++i) acc[nt][i] = 0.0f;

        #pragma unroll
        for (int kc = 0; kc < 5; ++kc) {
            v16h a;
            #pragma unroll
            for (int v = 0; v < 8; ++v) {
                const int kb = 2 * v + 8 * g + ((v >= 4) ? 8 : 0);
                const int kk = kc * 32 + kb;
                a[2 * v]     = (_Float16)fe_feat(kk,     x, fr);
                a[2 * v + 1] = (_Float16)fe_feat(kk + 1, x, fr);
            }
            const _Float16* wb = ws + ((size_t)(d * 5 + kc) * 8) * TILE_ELTS + lane * 16;
            #pragma unroll
            for (int nt = 0; nt < 8; ++nt) {   // 8 contiguous tiles -> clause-friendly b128 loads
                v16h b = *(const v16h*)(wb + (size_t)nt * TILE_ELTS);
                acc[nt] = __builtin_amdgcn_wmma_f32_16x16x32_f16(
                    false, a, false, b, (short)0, acc[nt], false, false);
            }
        }

        // --- bias + LayerNorm + ReLU (row = v + 8g, per-row reduce over 8 nt x 16 lanes) ---
        float bb[8], gg[8], be[8];
        #pragma unroll
        for (int nt = 0; nt < 8; ++nt) {
            const int h = nt * 16 + col;
            bb[nt] = b1[d * FE_H + h];
            gg[nt] = g1[d * FE_H + h];
            be[nt] = be1[d * FE_H + h];
        }
        #pragma unroll
        for (int v = 0; v < 8; ++v) {
            float s = 0.0f, s2 = 0.0f;
            #pragma unroll
            for (int nt = 0; nt < 8; ++nt) {
                const float t = acc[nt][v] + bb[nt];
                s += t; s2 += t * t;
            }
            #pragma unroll
            for (int m = 1; m < 16; m <<= 1) {
                s  += __shfl_xor(s,  m, 32);
                s2 += __shfl_xor(s2, m, 32);
            }
            const float mean = s * (1.0f / 128.0f);
            const float var  = s2 * (1.0f / 128.0f) - mean * mean;
            const float rstd = rsqrtf(var + FE_EPS);
            #pragma unroll
            for (int nt = 0; nt < 8; ++nt) {
                float t = (acc[nt][v] + bb[nt] - mean) * rstd * gg[nt] + be[nt];
                hbuf[wave][v + 8 * g][nt * 16 + col] = (_Float16)fmaxf(t, 0.0f);
            }
        }
        __builtin_amdgcn_wave_barrier();
        asm volatile("s_wait_dscnt 0" ::: "memory");   // LDS in-order per wave; be explicit

        // --- re-layout C->A via LDS, GEMM2 accumulate into acc2 (K-chunk outer) ---
        #pragma unroll
        for (int kc = 0; kc < 4; ++kc) {
            v16h a;
            #pragma unroll
            for (int v = 0; v < 8; ++v) {
                const int kb = 2 * v + 8 * g + ((v >= 4) ? 8 : 0);
                const int kk = kc * 32 + kb;
                a[2 * v]     = hbuf[wave][col][kk];
                a[2 * v + 1] = hbuf[wave][col][kk + 1];
            }
            const _Float16* wb = ws + ((size_t)(T2_BASE + (d * 4 + kc) * 8)) * TILE_ELTS + lane * 16;
            #pragma unroll
            for (int nt = 0; nt < 8; ++nt) {
                v16h b = *(const v16h*)(wb + (size_t)nt * TILE_ELTS);
                acc2[nt] = __builtin_amdgcn_wmma_f32_16x16x32_f16(
                    false, a, false, b, (short)0, acc2[nt], false, false);
            }
        }
        __builtin_amdgcn_wave_barrier();
    }

    // --- out head: +sum_d b2, LayerNorm, ReLU, GEMM3, +b3 ---
    float b2s[8], og8[8], ob8[8], b38[8];
    #pragma unroll
    for (int nt = 0; nt < 8; ++nt) {
        const int h = nt * 16 + col;
        b2s[nt] = b2[h] + b2[FE_H + h] + b2[2 * FE_H + h] + b2[3 * FE_H + h];
        og8[nt] = og[h]; ob8[nt] = ob[h]; b38[nt] = b3[h];
    }
    #pragma unroll
    for (int v = 0; v < 8; ++v) {
        float s = 0.0f, s2 = 0.0f;
        #pragma unroll
        for (int nt = 0; nt < 8; ++nt) {
            const float t = acc2[nt][v] + b2s[nt];
            s += t; s2 += t * t;
        }
        #pragma unroll
        for (int m = 1; m < 16; m <<= 1) {
            s  += __shfl_xor(s,  m, 32);
            s2 += __shfl_xor(s2, m, 32);
        }
        const float mean = s * (1.0f / 128.0f);
        const float var  = s2 * (1.0f / 128.0f) - mean * mean;
        const float rstd = rsqrtf(var + FE_EPS);
        #pragma unroll
        for (int nt = 0; nt < 8; ++nt) {
            float t = (acc2[nt][v] + b2s[nt] - mean) * rstd * og8[nt] + ob8[nt];
            hbuf[wave][v + 8 * g][nt * 16 + col] = (_Float16)fmaxf(t, 0.0f);
        }
    }
    __builtin_amdgcn_wave_barrier();
    asm volatile("s_wait_dscnt 0" ::: "memory");

    // --- GEMM3: nt outer (acc2 dead; a3[4] + single accumulator keeps pressure low) ---
    v16h a3[4];
    #pragma unroll
    for (int kc = 0; kc < 4; ++kc) {
        #pragma unroll
        for (int v = 0; v < 8; ++v) {
            const int kb = 2 * v + 8 * g + ((v >= 4) ? 8 : 0);
            const int kk = kc * 32 + kb;
            a3[kc][2 * v]     = hbuf[wave][col][kk];
            a3[kc][2 * v + 1] = hbuf[wave][col][kk + 1];
        }
    }
    #pragma unroll
    for (int nt = 0; nt < 8; ++nt) {
        v8f acc3;
        #pragma unroll
        for (int i = 0; i < 8; ++i) acc3[i] = 0.0f;
        const _Float16* wb = ws + ((size_t)T3_BASE + nt) * TILE_ELTS + lane * 16;
        #pragma unroll
        for (int kc = 0; kc < 4; ++kc) {
            v16h b = *(const v16h*)(wb + (size_t)kc * 8 * TILE_ELTS);
            acc3 = __builtin_amdgcn_wmma_f32_16x16x32_f16(
                false, a3[kc], false, b, (short)0, acc3, false, false);
        }
        #pragma unroll
        for (int v = 0; v < 8; ++v)
            out[(size_t)(row0 + v + 8 * g) * FE_H + nt * 16 + col] = acc3[v] + b38[nt];
    }
}

extern "C" void kernel_launch(void* const* d_in, const int* in_sizes, int n_in,
                              void* d_out, int out_size, void* d_ws, size_t ws_size,
                              hipStream_t stream) {
    const float* cin   = (const float*)d_in[0];
    const float* freqs = (const float*)d_in[1];
    const float* w1    = (const float*)d_in[2];
    const float* b1    = (const float*)d_in[3];
    const float* g1    = (const float*)d_in[4];
    const float* be1   = (const float*)d_in[5];
    const float* w2    = (const float*)d_in[6];
    const float* b2    = (const float*)d_in[7];
    const float* og    = (const float*)d_in[8];
    const float* ob    = (const float*)d_in[9];
    const float* w3    = (const float*)d_in[10];
    const float* b3    = (const float*)d_in[11];
    _Float16* ws = (_Float16*)d_ws;
    float*    out = (float*)d_out;

    const int N = in_sizes[0] / FE_D;          // 131072
    fe_prep<<<T1_COUNT + T2_COUNT + T3_COUNT, 32, 0, stream>>>(w1, w2, w3, ws);
    fe_main<<<N / 128, 256, 0, stream>>>(cin, freqs, b1, g1, be1, b2, og, ob, b3, ws, out);
    (void)n_in; (void)out_size; (void)ws_size;
}
